// Attention_40252433498402
// MI455X (gfx1250) — matrix operand order repaired and checked by reading, not compiled, not run on hardware
//
#include <hip/hip_runtime.h>

// ---------------------------------------------------------------------------
// Attention block for MI455X (gfx1250, wave32, WMMA f32_16x16x32_f16,
// async global->LDS DMA with ASYNCcnt double buffering)
// B=2, N=2048, dim=2048, HEADS=16, DIM_HEAD=128
// ---------------------------------------------------------------------------

typedef __attribute__((ext_vector_type(16))) _Float16 v16h;
typedef __attribute__((ext_vector_type(4)))  _Float16 v4h;
typedef __attribute__((ext_vector_type(8)))  float    v8f;

#define ROWS   4096      // B*N
#define DIM    2048
#define HEADS  16
#define DHEAD  128
#define NKEYS  2048      // N (keys per batch)

__device__ __forceinline__ v8f wmma_f16(v16h a, v16h b, v8f c) {
  // D = A(16x32 f16) * B(32x16 f16) + C(16x16 f32)
  return __builtin_amdgcn_wmma_f32_16x16x32_f16(
      /*neg_a=*/false, a, /*neg_b=*/false, b,
      /*c_mod=*/(short)0, c, /*reuse_a=*/false, /*reuse_b=*/false);
}

// A-fragment K index for element e (ISA 7.12.2, 16-bit A 16x32):
// lanes 0-15 hold K {0..7,16..23}; lanes 16-31 hold K {8..15,24..31}
__device__ __forceinline__ int a_kmap(int e, int half) {
  return ((e < 8) ? e : e + 8) + half * 8;
}

// Async global->LDS DMA (GLOBAL_LOAD_ASYNC_TO_LDS_B128, tracked by ASYNCcnt).
// LDS destination = low 32 bits of the generic pointer (flat->LDS aperture
// mapping).
__device__ __forceinline__ void async_load_b128(const void* g, void* l) {
  asm volatile("global_load_async_to_lds_b128 %0, %1, off"
               :: "v"((unsigned int)(size_t)l),
                  "v"((unsigned long long)(size_t)g)
               : "memory");
}
__device__ __forceinline__ void wait_async0() {
  asm volatile("s_wait_asynccnt 0x0" ::: "memory");
}

// ---------------------------------------------------------------------------
// Elementwise f32 -> f16 cast (weights)
// ---------------------------------------------------------------------------
__global__ __launch_bounds__(256)
void cast_to_f16(const float* __restrict__ X, _Float16* __restrict__ Y, int n) {
  int i = blockIdx.x * 256 + threadIdx.x;
  if (i < n) Y[i] = (_Float16)X[i];
}

// ---------------------------------------------------------------------------
// Block-wide sum (up to 8 waves of 32)
// ---------------------------------------------------------------------------
__device__ __forceinline__ float block_sum(float v, float* sbuf, int nwaves) {
  #pragma unroll
  for (int m = 16; m; m >>= 1) v += __shfl_xor(v, m, 32);
  int w = threadIdx.x >> 5;
  if ((threadIdx.x & 31) == 0) sbuf[w] = v;
  __syncthreads();
  float t = 0.f;
  for (int i = 0; i < nwaves; ++i) t += sbuf[i];
  __syncthreads();
  return t;
}

// ---------------------------------------------------------------------------
// LayerNorm over 2048 cols; one block per row
// ---------------------------------------------------------------------------
__global__ __launch_bounds__(256)
void ln_to_f16(const float* __restrict__ X, const float* __restrict__ g,
               _Float16* __restrict__ Y) {
  __shared__ float sbuf[8];
  const size_t row = blockIdx.x;
  const float* x = X + row * DIM;
  float loc[8], s = 0.f, s2 = 0.f;
  #pragma unroll
  for (int i = 0; i < 8; ++i) {
    float v = x[threadIdx.x + i * 256];
    loc[i] = v; s += v; s2 += v * v;
  }
  s  = block_sum(s,  sbuf, 8);
  s2 = block_sum(s2, sbuf, 8);
  float mu  = s  * (1.f / DIM);
  float var = s2 * (1.f / DIM) - mu * mu;
  float inv = rsqrtf(var + 1e-5f);
  #pragma unroll
  for (int i = 0; i < 8; ++i) {
    int c = threadIdx.x + i * 256;
    Y[row * DIM + c] = (_Float16)((loc[i] - mu) * inv * g[c]);
  }
}

__global__ __launch_bounds__(256)
void ln_to_f32(const float* __restrict__ X, const float* __restrict__ g,
               float* __restrict__ Y) {
  __shared__ float sbuf[8];
  const size_t row = blockIdx.x;
  const float* x = X + row * DIM;
  float loc[8], s = 0.f, s2 = 0.f;
  #pragma unroll
  for (int i = 0; i < 8; ++i) {
    float v = x[threadIdx.x + i * 256];
    loc[i] = v; s += v; s2 += v * v;
  }
  s  = block_sum(s,  sbuf, 8);
  s2 = block_sum(s2, sbuf, 8);
  float mu  = s  * (1.f / DIM);
  float var = s2 * (1.f / DIM) - mu * mu;
  float inv = rsqrtf(var + 1e-5f);
  #pragma unroll
  for (int i = 0; i < 8; ++i) {
    int c = threadIdx.x + i * 256;
    Y[row * DIM + c] = (loc[i] - mu) * inv * g[c];
  }
}

// ---------------------------------------------------------------------------
// WMMA GEMM: C(MxN,f32) = A(MxK,f16 rowmaj) * B(KxN,f16 rowmaj)
// 128x128 block tile, 256 threads (8 waves in 4Mx2N), K stepped by 32.
// Each wave: 32x64 -> 2 A-frags, 4 B-frags, 8 WMMAs per K-step.
// Double-buffered LDS filled by async global->LDS DMA.
// ---------------------------------------------------------------------------
__global__ __launch_bounds__(256)
void gemm_f16_f32(const _Float16* __restrict__ A, const _Float16* __restrict__ B,
                  float* __restrict__ C, int M, int N, int K) {
  __shared__ alignas(32) _Float16 As[2][128][40];    // padded (32 -> 40)
  __shared__ alignas(32) _Float16 Bs[2][32][136];    // padded (128 -> 136)
  const int tid  = threadIdx.x;
  const int wave = tid >> 5, lane = tid & 31;
  const int half = lane >> 4, ml = lane & 15;
  const int wm = (wave >> 1) * 32;                   // wave M offset 0/32/64/96
  const int wn = (wave & 1) * 64;                    // wave N offset 0/64
  const int bm = blockIdx.y * 128, bn = blockIdx.x * 128;

  v8f acc[2][4] = {};

  // Issue async DMA for one K-step tile pair: A 128x32, B 32x128 halves.
  // 512 b128 chunks each; 256 threads x 2 chunks per tile.
  auto issue_loads = [&](int buf, int k0) {
    #pragma unroll
    for (int it = 0; it < 2; ++it) {
      int idx = tid + it * 256;
      int r = idx >> 2, c = (idx & 3) * 8;           // 4 chunks per A row
      async_load_b128(&A[(size_t)(bm + r) * K + k0 + c], &As[buf][r][c]);
    }
    #pragma unroll
    for (int it = 0; it < 2; ++it) {
      int idx = tid + it * 256;
      int r = idx >> 4, c = (idx & 15) * 8;          // 16 chunks per B row
      async_load_b128(&B[(size_t)(k0 + r) * N + bn + c], &Bs[buf][r][c]);
    }
  };

  issue_loads(0, 0);
  wait_async0();
  __syncthreads();

  for (int k0 = 0, i = 0; k0 < K; k0 += 32, ++i) {
    const int cur = i & 1;
    if (k0 + 32 < K) issue_loads(cur ^ 1, k0 + 32);  // prefetch next K-step

    v16h a[2], b[4];
    #pragma unroll
    for (int mi = 0; mi < 2; ++mi) {
      const int m = wm + mi * 16 + ml;
      #pragma unroll
      for (int e = 0; e < 16; ++e) a[mi][e] = As[cur][m][a_kmap(e, half)];
    }
    #pragma unroll
    for (int n = 0; n < 4; ++n)
      #pragma unroll
      for (int e = 0; e < 16; ++e)
        b[n][e] = Bs[cur][a_kmap(e, half)][wn + n * 16 + ml];

    #pragma unroll
    for (int mi = 0; mi < 2; ++mi)
      #pragma unroll
      for (int n = 0; n < 4; ++n)
        acc[mi][n] = wmma_f16(a[mi], b[n], acc[mi][n]);

    // One wait+barrier covers both: next buffer DMA complete, and every wave
    // is done reading the current buffer before it gets overwritten.
    wait_async0();
    __syncthreads();
  }

  // C layout: lane 0-15 => N=lane, M=r; lane 16-31 => N=lane-16, M=r+8
  #pragma unroll
  for (int mi = 0; mi < 2; ++mi)
    #pragma unroll
    for (int n = 0; n < 4; ++n)
      #pragma unroll
      for (int r = 0; r < 8; ++r)
        C[(size_t)(bm + wm + mi * 16 + half * 8 + r) * N + bn + wn + n * 16 + ml]
            = acc[mi][n][r];
}

// ---------------------------------------------------------------------------
// q: per (token,head) -> qhat = l2norm(16*q) * 4, f16.  256 thr / token row.
// ---------------------------------------------------------------------------
__global__ __launch_bounds__(256)
void q_normalize(const float* __restrict__ Q, _Float16* __restrict__ Qh) {
  const size_t row = blockIdx.x;
  const int t = threadIdx.x;
  const int head = t >> 4;                 // 16 threads per head
  const int i0 = (t & 15) * 8;
  const float* q = Q + row * DIM + head * DHEAD;
  float loc[8], s2 = 0.f;
  #pragma unroll
  for (int i = 0; i < 8; ++i) {
    float v = q[i0 + i] * 16.0f;           // SCALE
    loc[i] = v; s2 += v * v;
  }
  #pragma unroll
  for (int m = 8; m; m >>= 1) s2 += __shfl_xor(s2, m, 16);
  float inv = rsqrtf(s2 + 1e-12f) * 4.0f;  // * sqrt(SCALE)
  _Float16* o = Qh + row * DIM + head * DHEAD + i0;
  #pragma unroll
  for (int i = 0; i < 8; ++i) o[i] = (_Float16)(loc[i] * inv);
}

// ---------------------------------------------------------------------------
// kv: khat = l2norm(k)*4 (f16), v -> f16.  128 thr / token row.
// ---------------------------------------------------------------------------
__global__ __launch_bounds__(128)
void kv_normalize(const float* __restrict__ KV, _Float16* __restrict__ Kh,
                  _Float16* __restrict__ Vh) {
  __shared__ float sb[4];
  const size_t row = blockIdx.x;
  const int t = threadIdx.x;
  float kv = KV[row * 256 + t];
  float vv = KV[row * 256 + 128 + t];
  float s2 = kv * kv;
  #pragma unroll
  for (int m = 16; m; m >>= 1) s2 += __shfl_xor(s2, m, 32);
  if ((t & 31) == 0) sb[t >> 5] = s2;
  __syncthreads();
  s2 = sb[0] + sb[1] + sb[2] + sb[3];
  float inv = rsqrtf(s2 + 1e-12f) * 4.0f;
  Kh[row * DHEAD + t] = (_Float16)(kv * inv);
  Vh[row * DHEAD + t] = (_Float16)vv;
}

// ---------------------------------------------------------------------------
// Flash attention: grid(N/64, B*H), 128 threads (4 waves).
// Wave owns 16 query rows; key tiles of 64; d=128 contracted in 4 WMMA steps.
// V tile loaded with async global->LDS DMA; K tile manually transposed.
// ---------------------------------------------------------------------------
__global__ __launch_bounds__(128)
void flash_attn(const _Float16* __restrict__ Qh, const _Float16* __restrict__ Kh,
                const _Float16* __restrict__ Vh, _Float16* __restrict__ Oh) {
  __shared__ alignas(32) _Float16 KsT[128][72];   // k^T : [d][key]
  __shared__ alignas(32) _Float16 Vs[64][136];    // [key][d], padded
  __shared__ alignas(32) _Float16 Ps[4][16][72];  // per-wave P tile (C->A relayout)
  const int tid = threadIdx.x, wave = tid >> 5, lane = tid & 31;
  const int half = lane >> 4, ml = lane & 15;
  const int b = blockIdx.y >> 4, h = blockIdx.y & 15;
  const int q0 = blockIdx.x * 64;
  const size_t rowbase = (size_t)b * NKEYS;       // token-row base of batch b

  // Q fragments (A layout), once per block
  v16h qf[4];
  {
    const size_t qrow = (rowbase + q0 + wave * 16 + ml) * DIM + (size_t)h * DHEAD;
    #pragma unroll
    for (int kc = 0; kc < 4; ++kc)
      #pragma unroll
      for (int e = 0; e < 16; ++e)
        qf[kc][e] = Qh[qrow + kc * 32 + a_kmap(e, half)];
  }

  v8f o[8] = {};
  float mrun[8], lrun[8];
  #pragma unroll
  for (int r = 0; r < 8; ++r) { mrun[r] = -3.0e38f; lrun[r] = 0.f; }

  for (int j0 = 0; j0 < NKEYS; j0 += 64) {
    // V tile: async DMA, 64x128 halves = 1024 b128 chunks / 128 threads
    #pragma unroll
    for (int it = 0; it < 8; ++it) {
      int idx = tid + it * 128;
      int key = idx >> 4, d = (idx & 15) * 8;
      async_load_b128(&Vh[(rowbase + j0 + key) * DHEAD + d], &Vs[key][d]);
    }
    // K tile transposed into LDS (manual; DMA cannot transpose)
    #pragma unroll
    for (int it = 0; it < 16; ++it) {
      int idx = (tid + it * 128) * 4;
      int key = idx >> 7, d = idx & 127;
      const _Float16* ks = &Kh[(rowbase + j0 + key) * DHEAD + d];
      KsT[d + 0][key] = ks[0]; KsT[d + 1][key] = ks[1];
      KsT[d + 2][key] = ks[2]; KsT[d + 3][key] = ks[3];
    }
    wait_async0();
    __syncthreads();

    // S = q @ k^T  (4 accumulators of 16 keys each)
    v8f s[4] = {};
    #pragma unroll
    for (int kc = 0; kc < 4; ++kc) {
      #pragma unroll
      for (int n = 0; n < 4; ++n) {
        v16h bfr;
        #pragma unroll
        for (int e = 0; e < 16; ++e)
          bfr[e] = KsT[kc * 32 + a_kmap(e, half)][n * 16 + ml];
        s[n] = wmma_f16(qf[kc], bfr, s[n]);
      }
    }

    // Online softmax (rows of C-frag live in one 16-lane half)
    #pragma unroll
    for (int r = 0; r < 8; ++r) {
      float mx = fmaxf(fmaxf(s[0][r], s[1][r]), fmaxf(s[2][r], s[3][r]));
      #pragma unroll
      for (int msk = 8; msk; msk >>= 1) mx = fmaxf(mx, __shfl_xor(mx, msk, 16));
      float mnew = fmaxf(mrun[r], mx);
      float sc = __expf(mrun[r] - mnew);
      lrun[r] *= sc;
      #pragma unroll
      for (int df = 0; df < 8; ++df) o[df][r] *= sc;
      mrun[r] = mnew;
    }
    #pragma unroll
    for (int r = 0; r < 8; ++r) {
      float rs = 0.f;
      #pragma unroll
      for (int n = 0; n < 4; ++n) {
        float p = __expf(s[n][r] - mrun[r]);
        rs += p;
        Ps[wave][half * 8 + r][n * 16 + ml] = (_Float16)p;   // C-layout store
      }
      #pragma unroll
      for (int msk = 8; msk; msk >>= 1) rs += __shfl_xor(rs, msk, 16);
      lrun[r] += rs;
    }
    __syncthreads();   // make cross-lane P visible before A-layout reads

    // O += P @ V
    #pragma unroll
    for (int kk = 0; kk < 2; ++kk) {
      v16h a;
      #pragma unroll
      for (int e = 0; e < 16; ++e)
        a[e] = Ps[wave][ml][kk * 32 + a_kmap(e, half)];
      #pragma unroll
      for (int df = 0; df < 8; ++df) {
        v16h bfr;
        #pragma unroll
        for (int e = 0; e < 16; ++e)
          bfr[e] = Vs[kk * 32 + a_kmap(e, half)][df * 16 + ml];
        o[df] = wmma_f16(a, bfr, o[df]);
      }
    }
    __syncthreads();
  }

  // Epilogue: O /= l, write f16 at (token row, h*128 + d)
  #pragma unroll
  for (int r = 0; r < 8; ++r) {
    float inv = 1.0f / lrun[r];
    size_t row = (rowbase + q0 + wave * 16 + half * 8 + r) * DIM + (size_t)h * DHEAD;
    #pragma unroll
    for (int df = 0; df < 8; ++df)
      Oh[row + df * 16 + ml] = (_Float16)(o[df][r] * inv);
  }
}

// ---------------------------------------------------------------------------
// Host-side orchestration
// ---------------------------------------------------------------------------
extern "C" void kernel_launch(void* const* d_in, const int* in_sizes, int n_in,
                              void* d_out, int out_size, void* d_ws, size_t ws_size,
                              hipStream_t stream) {
  const float* x      = (const float*)d_in[0];
  const float* g_norm = (const float*)d_in[1];
  const float* Wq     = (const float*)d_in[2];
  const float* Wkv    = (const float*)d_in[3];
  const float* Wout   = (const float*)d_in[4];
  const float* g_out  = (const float*)d_in[5];
  float* out = (float*)d_out;

  char* ws = (char*)d_ws;
  size_t off = 0;
  auto alloc = [&](size_t bytes) -> void* {
    void* p = ws + off;
    off = (off + bytes + 255) & ~(size_t)255;
    return p;
  };

  _Float16* xn     = (_Float16*)alloc((size_t)ROWS * DIM * 2);      // 16 MB
  _Float16* wq16   = (_Float16*)alloc((size_t)DIM * DIM * 2);       //  8 MB
  _Float16* wkv16  = (_Float16*)alloc((size_t)DIM * 256 * 2);       //  1 MB
  _Float16* wout16 = (_Float16*)alloc((size_t)DIM * DIM * 2);       //  8 MB
  float*    qf32   = (float*)   alloc((size_t)ROWS * DIM * 4);      // 32 MB (reused for out-proj)
  float*    kvf32  = (float*)   alloc((size_t)ROWS * 256 * 4);      //  4 MB
  _Float16* qh     = (_Float16*)alloc((size_t)ROWS * DIM * 2);      // 16 MB
  _Float16* kh     = (_Float16*)alloc((size_t)ROWS * DHEAD * 2);    //  1 MB
  _Float16* vh     = (_Float16*)alloc((size_t)ROWS * DHEAD * 2);    //  1 MB
  _Float16* oh     = (_Float16*)alloc((size_t)ROWS * DIM * 2);      // 16 MB

  // Weight casts
  cast_to_f16<<<(DIM * DIM + 255) / 256, 256, 0, stream>>>(Wq, wq16, DIM * DIM);
  cast_to_f16<<<(DIM * 256 + 255) / 256, 256, 0, stream>>>(Wkv, wkv16, DIM * 256);
  cast_to_f16<<<(DIM * DIM + 255) / 256, 256, 0, stream>>>(Wout, wout16, DIM * DIM);

  // LayerNorm(x) -> f16
  ln_to_f16<<<ROWS, 256, 0, stream>>>(x, g_norm, xn);

  // Projections (128x128 tiles)
  gemm_f16_f32<<<dim3(DIM / 128, ROWS / 128), 256, 0, stream>>>(xn, wq16, qf32, ROWS, DIM, DIM);
  gemm_f16_f32<<<dim3(256 / 128, ROWS / 128), 256, 0, stream>>>(xn, wkv16, kvf32, ROWS, 256, DIM);

  // Normalizations
  q_normalize <<<ROWS, 256, 0, stream>>>(qf32, qh);
  kv_normalize<<<ROWS, 128, 0, stream>>>(kvf32, kh, vh);

  // Flash attention: grid(N/64, B*HEADS)
  flash_attn<<<dim3(NKEYS / 64, 2 * HEADS), 128, 0, stream>>>(qh, kh, vh, oh);

  // Output projection (reuse qf32 as f32 scratch) + final LayerNorm
  gemm_f16_f32<<<dim3(DIM / 128, ROWS / 128), 256, 0, stream>>>(oh, wout16, qf32, ROWS, DIM, DIM);
  ln_to_f32<<<ROWS, 256, 0, stream>>>(qf32, g_out, out);
}
